// LSTM_90280212562189
// MI455X (gfx1250) — matrix operand order repaired and checked by reading, not compiled, not run on hardware
//
#include <hip/hip_runtime.h>
#include <stdint.h>

// ---------------------------------------------------------------------------
// Types for CDNA5 WMMA
// ---------------------------------------------------------------------------
typedef __attribute__((ext_vector_type(16))) __bf16 v16bf;
typedef __attribute__((ext_vector_type(8)))  float  v8f;
typedef __attribute__((ext_vector_type(4)))  unsigned int v4u;

union FragBF16 { v16bf v; v4u q[2]; };

// Problem constants
#define T_LEN  128
#define BATCH  32
#define ROWS   4096          // T_LEN * BATCH
#define NHID   512
#define G4H    2048          // 4 * NHID
#define NTOK   32000

__device__ __forceinline__ uint16_t f2bf(float x) {
    uint32_t u = __float_as_uint(x);
    u += 0x7FFFu + ((u >> 16) & 1u);     // round-to-nearest-even
    return (uint16_t)(u >> 16);
}
__device__ __forceinline__ float sigm(float x) { return 1.0f / (1.0f + __expf(-x)); }

// ---------------------------------------------------------------------------
// Embedding gather + fp32 -> bf16
// ---------------------------------------------------------------------------
__global__ __launch_bounds__(256) void embed_gather_bf16(
    const int* __restrict__ ids, const float* __restrict__ emb,
    uint16_t* __restrict__ xbf)
{
    int i = blockIdx.x * 256 + threadIdx.x;     // ROWS*NHID threads
    int r = i >> 9, k = i & 511;
    xbf[i] = f2bf(emb[(size_t)ids[r] * NHID + k]);
}

__global__ __launch_bounds__(256) void cast_f32_bf16(
    const float* __restrict__ in, uint16_t* __restrict__ out, int n)
{
    int i = blockIdx.x * 256 + threadIdx.x;
    if (i < n) out[i] = f2bf(in[i]);
}

// ---------------------------------------------------------------------------
// hterm[b][n] = b_ih[n] + b_hh[n] + dot(h_prev[b,:], W_hh[n,:])   (32 x 2048)
// ---------------------------------------------------------------------------
__global__ __launch_bounds__(256) void hterm_kernel(
    const float* __restrict__ hprev, const float* __restrict__ Whh,
    const float* __restrict__ bih, const float* __restrict__ bhh,
    float* __restrict__ ht)
{
    int i = blockIdx.x * 256 + threadIdx.x;     // BATCH*G4H threads
    int b = i >> 11, n = i & 2047;
    const float* hr = hprev + (size_t)b * NHID;
    const float* wr = Whh + (size_t)n * NHID;
    float s = bih[n] + bhh[n];
    #pragma unroll 4
    for (int k = 0; k < NHID; ++k) s += hr[k] * wr[k];
    ht[i] = s;
}

// ---------------------------------------------------------------------------
// bf16 WMMA GEMM:  C(M,N) = A(M,K) * B(N,K)^T
//   block = 256 threads = 8 waves in 2(M) x 4(N); block tile 64 x 256, BK=32
//   wave tile 32 x 64 = 8 accumulators; double-buffered LDS (1 barrier/iter)
//   mode 0: C[m*N+n] = acc                        (gate pre-activations)
//   mode 1: C[((m&31)*T + (m>>5))*N + n] = acc + bias[n]   (logits, permuted)
// ---------------------------------------------------------------------------
#define BK 32
#define LDS_STRIDE 40   // bf16 elements; row stride 80B keeps b128 alignment

__global__ __launch_bounds__(256) void gemm_wmma_bf16(
    const uint16_t* __restrict__ A, const uint16_t* __restrict__ B,
    float* __restrict__ C, const float* __restrict__ bias,
    int N, int K, int mode)
{
    __shared__ uint16_t As[2][64  * LDS_STRIDE];   // 2 x  5,120 B
    __shared__ uint16_t Bs[2][256 * LDS_STRIDE];   // 2 x 20,480 B  (51,200 B total)

    const int tid  = threadIdx.x;
    const int lane = tid & 31;
    const int wave = tid >> 5;
    const int wm   = wave & 1;       // 0..1  -> 32-row strip
    const int wn   = wave >> 1;      // 0..3  -> 64-col strip
    const int m0   = blockIdx.y * 64;
    const int n0   = blockIdx.x * 256;

    // staging: A 64x32 (one b128 per thread), B 256x32 (one row = 4 b128 per thread)
    const int arow = tid >> 2, acol = (tid & 3) * 8;

    // fragment LDS offsets (constant across K loop)
    const int l15 = lane & 15, lh = lane >> 4;
    int aoff[2], boff[4];
    #pragma unroll
    for (int i = 0; i < 2; ++i)
        aoff[i] = (wm * 32 + i * 16 + l15) * LDS_STRIDE + 8 * lh;
    #pragma unroll
    for (int j = 0; j < 4; ++j)
        boff[j] = (wn * 64 + j * 16 + l15) * LDS_STRIDE + 8 * lh;

    auto stage = [&](int k0, int buf) {
        const v4u* ga = (const v4u*)(A + (size_t)(m0 + arow) * K + k0 + acol);
        *(v4u*)(&As[buf][arow * LDS_STRIDE + acol]) = ga[0];
        const v4u* gb = (const v4u*)(B + (size_t)(n0 + tid) * K + k0);
        #pragma unroll
        for (int q = 0; q < 4; ++q)
            *(v4u*)(&Bs[buf][tid * LDS_STRIDE + q * 8]) = gb[q];
    };

    v8f acc[2][4] = {};
    stage(0, 0);
    int cur = 0;

    for (int k0 = 0; k0 < K; k0 += BK) {
        __syncthreads();                     // staging of `cur` complete
        if (k0 + BK < K) {
            stage(k0 + BK, cur ^ 1);         // overlap next tile with WMMAs
            if (k0 + 2 * BK < K) {           // global_prefetch_b8 of tile k+2
                __builtin_prefetch(A + (size_t)(m0 + arow) * K + k0 + 2 * BK + acol, 0, 1);
                __builtin_prefetch(B + (size_t)(n0 + tid) * K + k0 + 2 * BK, 0, 1);
            }
        }

        FragBF16 a[2], b[4];
        #pragma unroll
        for (int i = 0; i < 2; ++i) {
            a[i].q[0] = *(const v4u*)(&As[cur][aoff[i]]);       // K blocks 8h..8h+7
            a[i].q[1] = *(const v4u*)(&As[cur][aoff[i] + 16]);  // K blocks 16+8h..
        }
        #pragma unroll
        for (int j = 0; j < 4; ++j) {
            b[j].q[0] = *(const v4u*)(&Bs[cur][boff[j]]);       // K = 8h .. 8h+7
            b[j].q[1] = *(const v4u*)(&Bs[cur][boff[j] + 16]);  // K = 16+8h .. 16+8h+7
        }

        #pragma unroll
        for (int i = 0; i < 2; ++i)
            #pragma unroll
            for (int j = 0; j < 4; ++j)
                acc[i][j] = __builtin_amdgcn_wmma_f32_16x16x32_bf16(
                    false, a[i].v, false, b[j].v, (short)0, acc[i][j],
                    false, false);
        cur ^= 1;
    }

    // Epilogue. C/D layout: VGPR e -> M = e + 8*(lane>=16); N = lane & 15
    #pragma unroll
    for (int i = 0; i < 2; ++i) {
        #pragma unroll
        for (int j = 0; j < 4; ++j) {
            const int nn = n0 + wn * 64 + j * 16 + l15;
            const int mb = m0 + wm * 32 + i * 16 + 8 * lh;
            #pragma unroll
            for (int e = 0; e < 8; ++e) {
                const int m = mb + e;
                const float v = acc[i][j][e];
                if (mode == 0) {
                    C[(size_t)m * N + nn] = v;
                } else {
                    const int t = m >> 5, bb = m & 31;   // row r = t*BATCH + b
                    C[((size_t)(bb * T_LEN + t)) * N + nn] = v + bias[nn];
                }
            }
        }
    }
}

// ---------------------------------------------------------------------------
// LSTM gate nonlinearities. One thread per (row, hidden unit).
// ---------------------------------------------------------------------------
__global__ __launch_bounds__(256) void lstm_gates(
    const float* __restrict__ G, const float* __restrict__ ht,
    const float* __restrict__ cprev,
    float* __restrict__ hOut, float* __restrict__ cOut,
    uint16_t* __restrict__ hbf)
{
    int i = blockIdx.x * 256 + threadIdx.x;     // ROWS*NHID threads
    int r = i >> 9, j = i & 511, b = r & 31;
    const float* gr = G + (size_t)r * G4H;
    const float* hb = ht + (size_t)b * G4H;
    float gi = gr[j]        + hb[j];
    float gf = gr[512 + j]  + hb[512 + j];
    float gg = gr[1024 + j] + hb[1024 + j];
    float go = gr[1536 + j] + hb[1536 + j];
    float c2 = sigm(gf) * cprev[b * NHID + j] + sigm(gi) * tanhf(gg);
    float h2 = sigm(go) * tanhf(c2);
    hOut[i] = h2;
    cOut[i] = c2;
    hbf[i]  = f2bf(h2);
}

// ---------------------------------------------------------------------------
// In-place log_softmax over V=32000 per row; whole row cached in 128KB LDS
// (CDNA5: up to 320KB per workgroup).
// ---------------------------------------------------------------------------
__global__ __launch_bounds__(1024) void logsoftmax_rows(float* __restrict__ out)
{
    extern __shared__ float buf[];          // NTOK floats
    __shared__ float red[32];
    const int tid = threadIdx.x;
    float* rowp = out + (size_t)blockIdx.x * NTOK;

    float lmax = -3.4e38f;
    for (int i = tid; i < NTOK; i += 1024) {
        float v = rowp[i];
        buf[i] = v;
        lmax = fmaxf(lmax, v);
    }
    #pragma unroll
    for (int o = 16; o > 0; o >>= 1) lmax = fmaxf(lmax, __shfl_xor(lmax, o, 32));
    if ((tid & 31) == 0) red[tid >> 5] = lmax;
    __syncthreads();
    if (tid < 32) {
        float t = red[tid];
        #pragma unroll
        for (int o = 16; o > 0; o >>= 1) t = fmaxf(t, __shfl_xor(t, o, 32));
        if (tid == 0) red[0] = t;
    }
    __syncthreads();
    const float rmax = red[0];
    __syncthreads();

    float lsum = 0.0f;
    for (int i = tid; i < NTOK; i += 1024) lsum += __expf(buf[i] - rmax);
    #pragma unroll
    for (int o = 16; o > 0; o >>= 1) lsum += __shfl_xor(lsum, o, 32);
    if ((tid & 31) == 0) red[tid >> 5] = lsum;
    __syncthreads();
    if (tid < 32) {
        float t = red[tid];
        #pragma unroll
        for (int o = 16; o > 0; o >>= 1) t += __shfl_xor(t, o, 32);
        if (tid == 0) red[0] = t;
    }
    __syncthreads();
    const float shift = __logf(red[0]) + rmax;

    for (int i = tid; i < NTOK; i += 1024) rowp[i] = buf[i] - shift;
}

// ---------------------------------------------------------------------------
// Host launcher
// ---------------------------------------------------------------------------
extern "C" void kernel_launch(void* const* d_in, const int* in_sizes, int n_in,
                              void* d_out, int out_size, void* d_ws, size_t ws_size,
                              hipStream_t stream)
{
    (void)in_sizes; (void)n_in; (void)out_size; (void)ws_size;

    const int*   ids   = (const int*)  d_in[0];
    const float* h0    = (const float*)d_in[1];
    const float* c0    = (const float*)d_in[2];
    const float* h1    = (const float*)d_in[3];
    const float* c1    = (const float*)d_in[4];
    const float* emb   = (const float*)d_in[5];
    const float* Wih0  = (const float*)d_in[6];
    const float* Whh0  = (const float*)d_in[7];
    const float* bih0  = (const float*)d_in[8];
    const float* bhh0  = (const float*)d_in[9];
    const float* Wih1  = (const float*)d_in[10];
    const float* Whh1  = (const float*)d_in[11];
    const float* bih1  = (const float*)d_in[12];
    const float* bhh1  = (const float*)d_in[13];
    const float* decW  = (const float*)d_in[14];
    const float* decb  = (const float*)d_in[15];

    float* out = (float*)d_out;
    // d_out layout: logp (B,T,V) | hid_h (2,T,B,H) | hid_c (2,T,B,H)
    const size_t OFF_HIDH = (size_t)BATCH * T_LEN * NTOK;            // 131,072,000
    const size_t OFF_HIDC = OFF_HIDH + (size_t)2 * ROWS * NHID;      // +4,194,304
    const size_t LAYER    = (size_t)ROWS * NHID;                     // 2,097,152

    // Workspace carve-up (bytes)
    char* w = (char*)d_ws;
    uint16_t* Xbf     = (uint16_t*)(w);                        //  4,194,304 B
    uint16_t* Wih0bf  = (uint16_t*)(w + 4194304);              //  2,097,152 B
    uint16_t* Wih1bf  = (uint16_t*)(w + 6291456);              //  2,097,152 B
    uint16_t* decWbf  = (uint16_t*)(w + 8388608);              // 32,768,000 B
    float*    hterm0  = (float*)   (w + 41156608);             //    262,144 B
    float*    hterm1  = (float*)   (w + 41418752);             //    262,144 B
    float*    Gbuf    = (float*)   (w + 41680896);             // 33,554,432 B
    uint16_t* H0bf    = (uint16_t*)(w + 75235328);             //  4,194,304 B
    uint16_t* H1bf    = (uint16_t*)(w + 79429632);             //  4,194,304 B
    // total ~83.6 MB

    // 1. embedding gather -> bf16
    embed_gather_bf16<<<(ROWS * NHID) / 256, 256, 0, stream>>>(ids, emb, Xbf);

    // 2. weight conversions
    cast_f32_bf16<<<(G4H * NHID) / 256, 256, 0, stream>>>(Wih0, Wih0bf, G4H * NHID);
    cast_f32_bf16<<<(G4H * NHID) / 256, 256, 0, stream>>>(Wih1, Wih1bf, G4H * NHID);
    cast_f32_bf16<<<(NTOK * NHID) / 256, 256, 0, stream>>>(decW, decWbf, NTOK * NHID);

    // 3. broadcast h-terms (tiny)
    hterm_kernel<<<(BATCH * G4H) / 256, 256, 0, stream>>>(h0, Whh0, bih0, bhh0, hterm0);
    hterm_kernel<<<(BATCH * G4H) / 256, 256, 0, stream>>>(h1, Whh1, bih1, bhh1, hterm1);

    // 4. layer 0: G = X * Wih0^T   (4096 x 2048, K=512)
    gemm_wmma_bf16<<<dim3(G4H / 256, ROWS / 64), 256, 0, stream>>>(
        Xbf, Wih0bf, Gbuf, nullptr, G4H, NHID, 0);
    lstm_gates<<<(ROWS * NHID) / 256, 256, 0, stream>>>(
        Gbuf, hterm0, c0, out + OFF_HIDH, out + OFF_HIDC, H0bf);

    // 5. layer 1
    gemm_wmma_bf16<<<dim3(G4H / 256, ROWS / 64), 256, 0, stream>>>(
        H0bf, Wih1bf, Gbuf, nullptr, G4H, NHID, 0);
    lstm_gates<<<(ROWS * NHID) / 256, 256, 0, stream>>>(
        Gbuf, hterm1, c1, out + OFF_HIDH + LAYER, out + OFF_HIDC + LAYER, H1bf);

    // 6. decoder: logits (permuted to (B,T,V)) = H1 * decW^T + decb
    gemm_wmma_bf16<<<dim3(NTOK / 256, ROWS / 64), 256, 0, stream>>>(
        H1bf, decWbf, out, decb, NTOK, NHID, 1);

    // 7. in-place log_softmax over V per (b,t) row; row cached in LDS
    logsoftmax_rows<<<ROWS, 1024, NTOK * sizeof(float), stream>>>(out);
}
